// AtomTransformer_21852793602159
// MI455X (gfx1250) — matrix operand order repaired and checked by reading, not compiled, not run on hardware
//
#include <hip/hip_runtime.h>

// ---------------------------------------------------------------------------
// AtomTransformer (AF3 Alg 23) for MI455X / gfx1250, wave32 + WMMA bf16.
//
// ~4.2 GFLOP total, ~70 MB unique bytes (resident in 192MB L2 after block 0)
// -> compute/latency bound. All GEMMs + attention QK^T / PV use
// v_wmma_f32_16x16x32_bf16 (bf16 in, f32 accumulate). Weights are packed
// fragment-major at f32->bf16 conversion so every B-fragment is one
// contiguous 32B per-lane load (2x global_load_b128 instead of 16x u16).
// Attention exploits the _beta neighborhood: query block qb (32 rows) attends
// only to keys [32*qb-48, 32*qb+79]; V is stored transposed with a zeroed
// 64-key halo so PV fragments are contiguous too.
// ---------------------------------------------------------------------------

#define NATOM  2048
#define VPAD   64
#define NATOMP (NATOM + 2 * VPAD)   // 2176: key-padded stride for V^T
#define CQ     128      // c_q
#define CZP    16       // c_p
#define NHEAD  4
#define DHEAD  32
#define NBLK   3
#define NTC    256      // NT * C
#define NQW    32       // n_query
#define NKWIN  128      // n_key window
#define INF_F  1.0e9f
#define LN_EPS 1e-5f

typedef __bf16 bf16;
typedef __attribute__((ext_vector_type(16))) __bf16 v16bf;
typedef __attribute__((ext_vector_type(8)))  __bf16 v8bf;
typedef __attribute__((ext_vector_type(8)))  float  v8f;

union BfBits { unsigned short u; bf16 b; };

__device__ __forceinline__ bf16 f2bf(float f) {
  unsigned u = __float_as_uint(f);
  u += 0x7fffu + ((u >> 16) & 1u);          // round-to-nearest-even
  BfBits c; c.u = (unsigned short)(u >> 16);
  return c.b;
}
__device__ __forceinline__ float sigf(float x) { return 1.0f / (1.0f + __expf(-x)); }

__device__ __forceinline__ v8f vzero8() {
  v8f z = {0.f, 0.f, 0.f, 0.f, 0.f, 0.f, 0.f, 0.f};
  return z;
}

// --- WMMA fragment loaders (contiguous k order k = 16*(lane/16) + e in A and B) -------
// A fragment: 16(M) x 32(K) bf16, row-major source, row stride lda.
// Per lane: two contiguous 16B chunks (vectorized to b128 loads).
__device__ __forceinline__ v16bf load_a_frag(const bf16* A, int lda, int lane) {
  const int m  = lane & 15;
  const int kh = (lane >> 4) << 4;   // 0 or 16
  const bf16* row = A + (size_t)m * lda + kh;
  v8bf lo = *(const v8bf*)(row);
  v8bf hi = *(const v8bf*)(row + 8);
  return __builtin_shufflevector(lo, hi, 0, 1, 2, 3, 4, 5, 6, 7,
                                 8, 9, 10, 11, 12, 13, 14, 15);
}

// B fragment from fragment-major packed weights: one contiguous 32B load/lane.
// Packed layout: dst[((kt*ntn + nt)*32 + lane)*16 + e] with
//   k = kt*32 + ((lane>>4)<<4) + e,  n = nt*16 + (lane&15).
__device__ __forceinline__ v16bf load_b_packed(const bf16* Bp, int tile, int lane) {
  return *(const v16bf*)(Bp + (((size_t)tile * 32 + lane) << 4));
}

__device__ __forceinline__ v8f wmma_bf16(v16bf a, v16bf b, v8f c) {
  return __builtin_amdgcn_wmma_f32_16x16x32_bf16(false, a, false, b, (short)0, c,
                                                 false, false);
}

// --- weight pack: f32 [K,N] row-major -> bf16 fragment-major ---------------
__global__ void pack_b_kernel(const float* __restrict__ src, bf16* __restrict__ dst,
                              int K, int N) {
  const int i = blockIdx.x * blockDim.x + threadIdx.x;
  if (i >= K * N) return;
  const int e    = i & 15;
  const int lane = (i >> 4) & 31;
  const int tile = i >> 9;
  const int ntn  = N >> 4;
  const int kt = tile / ntn, nt = tile % ntn;
  const int k = kt * 32 + ((lane >> 4) << 4) + e;
  const int n = nt * 16 + (lane & 15);
  dst[i] = f2bf(src[(size_t)k * N + n]);
}

// --- small elementwise kernels ---------------------------------------------
__global__ void cvt_bf16_kernel(const float* __restrict__ in, bf16* __restrict__ out, int n) {
  const int i = blockIdx.x * blockDim.x + threadIdx.x;
  if (i < n) out[i] = f2bf(in[i]);
}

__global__ void scale_bf16_kernel(const float* __restrict__ in, const float* __restrict__ g,
                                  bf16* __restrict__ out, int n) {
  const int i = blockIdx.x * blockDim.x + threadIdx.x;
  if (i < n) out[i] = f2bf(in[i] * g[i & (CQ - 1)]);
}

// Zero the 64-key halos of V^T (CQ columns x 128 halo keys).
__global__ void zero_vt_pad_kernel(bf16* __restrict__ Vt) {
  const int i = blockIdx.x * blockDim.x + threadIdx.x;
  if (i >= CQ * 2 * VPAD) return;
  const int col = i >> 7;
  const int j   = i & 127;
  const size_t idx = (size_t)col * NATOMP + (j < VPAD ? j : NATOM + j);
  BfBits z; z.u = 0;
  Vt[idx] = z.b;
}

// Row LayerNorm over CQ=128 channels (biased variance, eps=1e-5).
__global__ void ln_kernel(const float* __restrict__ in, float* __restrict__ out) {
  __shared__ float red[CQ];
  const int row = blockIdx.x, t = threadIdx.x;
  const float x = in[(size_t)row * CQ + t];
  red[t] = x;
  __syncthreads();
#pragma unroll
  for (int s = CQ / 2; s > 0; s >>= 1) {
    if (t < s) red[t] += red[t + s];
    __syncthreads();
  }
  const float mean = red[0] * (1.0f / CQ);
  __syncthreads();
  const float d = x - mean;
  red[t] = d * d;
  __syncthreads();
#pragma unroll
  for (int s = CQ / 2; s > 0; s >>= 1) {
    if (t < s) red[t] += red[t + s];
    __syncthreads();
  }
  const float var = red[0] * (1.0f / CQ);
  out[(size_t)row * CQ + t] = d * rsqrtf(var + LN_EPS);
}

// --- generic single GEMM: out = epi(A[M,K] @ B[K,N] + bias), B packed ------
#define EPI_PLAIN      0
#define EPI_SIGMUL     2   // sigmoid(acc+bias) * mul
#define EPI_SIGMUL_ADD 3   // sigmoid(acc+bias) * mul + add

template <int MODE>
__global__ void gemm_kernel(const bf16* __restrict__ A, const bf16* __restrict__ Bp,
                            const float* __restrict__ bias,
                            const float* __restrict__ mul, const float* __restrict__ add,
                            float* __restrict__ outF, int M, int N, int K) {
  const int lane = threadIdx.x & 31;
  const int wave = threadIdx.x >> 5;
  const int nt_n = N >> 4;
  const int tile = blockIdx.x * (blockDim.x >> 5) + wave;
  if (tile >= (M >> 4) * nt_n) return;
  const int mt = tile / nt_n, nt = tile % nt_n;
  const bf16* Ab = A + (size_t)(mt * 16) * K;
  v8f acc = vzero8();
  for (int k = 0; k < K; k += 32)
    acc = wmma_bf16(load_a_frag(Ab + k, K, lane),
                    load_b_packed(Bp, (k >> 5) * nt_n + nt, lane), acc);
  const int n = nt * 16 + (lane & 15);
  const float bv = bias ? bias[n] : 0.f;
  const int mbase = mt * 16 + ((lane >> 4) << 3);
#pragma unroll
  for (int r = 0; r < 8; ++r) {
    const size_t idx = (size_t)(mbase + r) * N + n;
    float v = acc[r] + bv;
    if (MODE == EPI_SIGMUL)          v = sigf(v) * mul[idx];
    else if (MODE == EPI_SIGMUL_ADD) v = sigf(v) * mul[idx] + add[idx];
    outF[idx] = v;
  }
}

// --- AdaLN dual GEMM: x = sigmoid(sn@sigW + sigb) * ln(a) + sn@skipW -> bf16
__global__ void adaln_dual_kernel(const bf16* __restrict__ A, const bf16* __restrict__ Bsig,
                                  const bf16* __restrict__ Bskip,
                                  const float* __restrict__ sigb, const float* __restrict__ lna,
                                  bf16* __restrict__ outX) {
  const int N = CQ, K = CQ;
  const int lane = threadIdx.x & 31;
  const int wave = threadIdx.x >> 5;
  const int nt_n = N >> 4;
  const int tile = blockIdx.x * (blockDim.x >> 5) + wave;
  if (tile >= (NATOM >> 4) * nt_n) return;
  const int mt = tile / nt_n, nt = tile % nt_n;
  const bf16* Ab = A + (size_t)(mt * 16) * K;
  v8f a1 = vzero8(), a2 = vzero8();
  for (int k = 0; k < K; k += 32) {
    v16bf af = load_a_frag(Ab + k, K, lane);
    const int bt = (k >> 5) * nt_n + nt;
    a1 = wmma_bf16(af, load_b_packed(Bsig, bt, lane), a1);
    a2 = wmma_bf16(af, load_b_packed(Bskip, bt, lane), a2);
  }
  const int n = nt * 16 + (lane & 15);
  const int mbase = mt * 16 + ((lane >> 4) << 3);
  const float bv = sigb[n];
#pragma unroll
  for (int r = 0; r < 8; ++r) {
    const size_t idx = (size_t)(mbase + r) * N + n;
    outX[idx] = f2bf(sigf(a1[r] + bv) * lna[idx] + a2[r]);
  }
}

// --- SwiGLU dual GEMM: hid = silu(x@w1) * (x@w2) -> bf16 [NATOM, 256] ------
__global__ void swiglu_dual_kernel(const bf16* __restrict__ A, const bf16* __restrict__ B1,
                                   const bf16* __restrict__ B2, bf16* __restrict__ outH) {
  const int N = NTC, K = CQ;
  const int lane = threadIdx.x & 31;
  const int wave = threadIdx.x >> 5;
  const int nt_n = N >> 4;
  const int tile = blockIdx.x * (blockDim.x >> 5) + wave;
  if (tile >= (NATOM >> 4) * nt_n) return;
  const int mt = tile / nt_n, nt = tile % nt_n;
  const bf16* Ab = A + (size_t)(mt * 16) * K;
  v8f a1 = vzero8(), a2 = vzero8();
  for (int k = 0; k < K; k += 32) {
    v16bf af = load_a_frag(Ab + k, K, lane);
    const int bt = (k >> 5) * nt_n + nt;
    a1 = wmma_bf16(af, load_b_packed(B1, bt, lane), a1);
    a2 = wmma_bf16(af, load_b_packed(B2, bt, lane), a2);
  }
  const int n = nt * 16 + (lane & 15);
  const int mbase = mt * 16 + ((lane >> 4) << 3);
#pragma unroll
  for (int r = 0; r < 8; ++r) {
    const size_t idx = (size_t)(mbase + r) * N + n;
    const float s = a1[r];
    outH[idx] = f2bf(s * sigf(s) * a2[r]);
  }
}

// --- Quad GEMM: Q(+bq), K (bf16 row-major), V (bf16 transposed+halo),
//                G = sigmoid(x@wg) (f32) -----------------------------------
__global__ void qkvg_kernel(const bf16* __restrict__ X,
                            const bf16* __restrict__ Wq, const float* __restrict__ bq,
                            const bf16* __restrict__ Wk, const bf16* __restrict__ Wv,
                            const bf16* __restrict__ Wg,
                            bf16* __restrict__ Q, bf16* __restrict__ Ko,
                            bf16* __restrict__ Vt, float* __restrict__ G) {
  const int N = CQ, K = CQ;
  const int lane = threadIdx.x & 31;
  const int wave = threadIdx.x >> 5;
  const int nt_n = N >> 4;
  const int tile = blockIdx.x * (blockDim.x >> 5) + wave;
  if (tile >= (NATOM >> 4) * nt_n) return;
  const int mt = tile / nt_n, nt = tile % nt_n;
  const bf16* Ab = X + (size_t)(mt * 16) * K;
  v8f aq = vzero8(), ak = vzero8(), av = vzero8(), ag = vzero8();
  for (int k = 0; k < K; k += 32) {
    v16bf af = load_a_frag(Ab + k, K, lane);
    const int bt = (k >> 5) * nt_n + nt;
    aq = wmma_bf16(af, load_b_packed(Wq, bt, lane), aq);
    ak = wmma_bf16(af, load_b_packed(Wk, bt, lane), ak);
    av = wmma_bf16(af, load_b_packed(Wv, bt, lane), av);
    ag = wmma_bf16(af, load_b_packed(Wg, bt, lane), ag);
  }
  const int n = nt * 16 + (lane & 15);
  const int mbase = mt * 16 + ((lane >> 4) << 3);
  const float bqv = bq[n];
#pragma unroll
  for (int r = 0; r < 8; ++r) {
    const int row = mbase + r;
    const size_t idx = (size_t)row * N + n;
    Q[idx]  = f2bf(aq[r] + bqv);
    Ko[idx] = f2bf(ak[r]);
    Vt[(size_t)n * NATOMP + VPAD + row] = f2bf(av[r]);   // transposed + halo
    G[idx]  = sigf(ag[r]);
  }
}

// --- Local pair-biased attention -------------------------------------------
// grid = (64 query blocks, 4 heads), 128 threads (4 wave32s).
// Query block qb attends to key window [32*qb-48, 32*qb+79].
__global__ void attn_kernel(const bf16* __restrict__ Q, const bf16* __restrict__ Km,
                            const bf16* __restrict__ Vt, const float* __restrict__ G,
                            const float* __restrict__ plm, const float* __restrict__ mask,
                            const float* __restrict__ zg, const float* __restrict__ zbv,
                            const float* __restrict__ wz, bf16* __restrict__ Og) {
  const int qb   = blockIdx.x;
  const int h    = blockIdx.y;
  const int tid  = threadIdx.x;
  const int lane = tid & 31;
  const int wave = tid >> 5;
  const int win0 = qb * NQW - 48;

  __shared__ float logits[NQW][NKWIN];                 // 16 KB
  __shared__ __align__(32) bf16 p_lds[NQW][NKWIN];     //  8 KB

  // Step 1: bias = (LN(plm)*zg+zb)@wz + key-mask; -INF outside window.
  for (int idx = tid; idx < NQW * NKWIN; idx += blockDim.x) {
    const int i  = idx >> 7;
    const int kk = idx & (NKWIN - 1);
    const int kg = win0 + kk;
    float b;
    if (kg < 0 || kg >= NATOM) {
      b = -INF_F;
    } else {
      const float* p = plm + ((size_t)(qb * NQW + i) * NATOM + kg) * CZP;
      __builtin_prefetch(p + (size_t)NATOM * CZP, 0, 1);   // next query row
      float mean = 0.f;
#pragma unroll
      for (int c = 0; c < CZP; ++c) mean += p[c];
      mean *= (1.0f / CZP);
      float var = 0.f;
#pragma unroll
      for (int c = 0; c < CZP; ++c) { const float d = p[c] - mean; var += d * d; }
      var *= (1.0f / CZP);
      const float rs = rsqrtf(var + LN_EPS);
      float z = 0.f;
#pragma unroll
      for (int c = 0; c < CZP; ++c)
        z += ((p[c] - mean) * rs * zg[c] + zbv[c]) * wz[c * NHEAD + h];
      b = z + (mask[kg] - 1.0f) * INF_F;
    }
    logits[i][kk] = b;
  }
  __syncthreads();

  // Step 2: logits += QK^T / sqrt(DH). 16 tiles (2 M x 8 keys), 4 per wave.
  const float scale = 0.17677669529663687f;  // 1/sqrt(32)
  for (int t = wave; t < 16; t += 4) {
    const int mi = t >> 3, ki = t & 7;
    v16bf af = load_a_frag(Q + (size_t)(qb * NQW + mi * 16) * CQ + h * DHEAD, CQ, lane);
    const int n  = lane & 15;
    const int k0 = (lane >> 4) << 4;
    int key = win0 + ki * 16 + n;
    key = key < 0 ? 0 : (key > NATOM - 1 ? NATOM - 1 : key);
    // K fragment: per-lane contiguous 16 bf16 from one key row (32B aligned).
    v16bf bfr = *(const v16bf*)(Km + (size_t)key * CQ + h * DHEAD + k0);
    v8f acc = wmma_bf16(af, bfr, vzero8());
    const int nn = ki * 16 + n;
    const int m0 = mi * 16 + ((lane >> 4) << 3);
#pragma unroll
    for (int r = 0; r < 8; ++r) logits[m0 + r][nn] += acc[r] * scale;
  }
  __syncthreads();

  // Step 3: row softmax (32 rows, one thread each).
  if (tid < NQW) {
    float mx = -3.0e38f;
    for (int k = 0; k < NKWIN; ++k) mx = fmaxf(mx, logits[tid][k]);
    float sum = 0.f;
    for (int k = 0; k < NKWIN; ++k) {
      const float e = __expf(logits[tid][k] - mx);
      logits[tid][k] = e;
      sum += e;
    }
    const float inv = 1.0f / sum;
    for (int k = 0; k < NKWIN; ++k) p_lds[tid][k] = f2bf(logits[tid][k] * inv);
  }
  __syncthreads();

  // Step 4: O = P @ V (per-wave 16x16 tile of the 32x32 head output), gated.
  {
    const int mi = wave >> 1, ni = wave & 1;
    const int n  = lane & 15;
    const int k0 = (lane >> 4) << 4;
    const int colv = h * DHEAD + ni * 16 + n;
    v8f acc = vzero8();
    for (int ks = 0; ks < NKWIN; ks += 32) {
      v16bf af = load_a_frag(&p_lds[mi * 16][ks], NKWIN, lane);
      // V^T fragment: contiguous 16 keys in one column; halo covers overhang.
      v16bf bfr = *(const v16bf*)(Vt + (size_t)colv * NATOMP + VPAD + win0 + ks + k0);
      acc = wmma_bf16(af, bfr, acc);
    }
    const int m0 = qb * NQW + mi * 16 + ((lane >> 4) << 3);
#pragma unroll
    for (int r = 0; r < 8; ++r) {
      const size_t idx = (size_t)(m0 + r) * CQ + colv;
      Og[idx] = f2bf(acc[r] * G[idx]);
    }
  }
}

// ---------------------------------------------------------------------------
extern "C" void kernel_launch(void* const* d_in, const int* in_sizes, int n_in,
                              void* d_out, int out_size, void* d_ws, size_t ws_size,
                              hipStream_t stream) {
  (void)in_sizes; (void)n_in; (void)out_size; (void)ws_size;
  const float* ql        = (const float*)d_in[0];
  const float* cl        = (const float*)d_in[1];
  const float* plm       = (const float*)d_in[2];
  const float* amask     = (const float*)d_in[3];
  const float* at_sln_g  = (const float*)d_in[4];
  const float* at_sig_w  = (const float*)d_in[5];
  const float* at_sig_b  = (const float*)d_in[6];
  const float* at_skip_w = (const float*)d_in[7];
  const float* at_wq     = (const float*)d_in[8];
  const float* at_bq     = (const float*)d_in[9];
  const float* at_wk     = (const float*)d_in[10];
  const float* at_wv     = (const float*)d_in[11];
  const float* at_zln_g  = (const float*)d_in[12];
  const float* at_zln_b  = (const float*)d_in[13];
  const float* at_wz     = (const float*)d_in[14];
  const float* at_wg     = (const float*)d_in[15];
  const float* at_wo     = (const float*)d_in[16];
  const float* at_ws_    = (const float*)d_in[17];
  const float* at_bs     = (const float*)d_in[18];
  const float* tr_sln_g  = (const float*)d_in[19];
  const float* tr_sig_w  = (const float*)d_in[20];
  const float* tr_sig_b  = (const float*)d_in[21];
  const float* tr_skip_w = (const float*)d_in[22];
  const float* tr_w1     = (const float*)d_in[23];
  const float* tr_w2     = (const float*)d_in[24];
  const float* tr_wo     = (const float*)d_in[25];
  const float* tr_ws_    = (const float*)d_in[26];
  const float* tr_bs     = (const float*)d_in[27];
  float* out = (float*)d_out;

  // ---- carve workspace ----
  char* ws = (char*)d_ws;
  size_t off = 0;
  auto carve = [&](size_t bytes) -> void* {
    void* p = ws + off;
    off += (bytes + 255) & ~(size_t)255;
    return p;
  };
  const size_t NC = (size_t)NATOM * CQ;
  const size_t CC = (size_t)CQ * CQ;
  float* a_cur   = (float*)carve(NC * 4);
  float* lncl    = (float*)carve(NC * 4);
  float* lna     = (float*)carve(NC * 4);
  float* gbuf    = (float*)carve(NC * 4);
  float* attnpre = (float*)carve(NC * 4);
  float* attnout = (float*)carve(NC * 4);
  float* tpre    = (float*)carve(NC * 4);
  bf16* cl_bf  = (bf16*)carve(NC * 2);
  bf16* sn_bf  = (bf16*)carve(NC * 2);
  bf16* x_bf   = (bf16*)carve(NC * 2);
  bf16* q_bf   = (bf16*)carve(NC * 2);
  bf16* k_bf   = (bf16*)carve(NC * 2);
  bf16* vt_bf  = (bf16*)carve((size_t)CQ * NATOMP * 2);   // V^T with key halo
  bf16* og_bf  = (bf16*)carve(NC * 2);
  bf16* xt_bf  = (bf16*)carve(NC * 2);
  bf16* hid_bf = (bf16*)carve((size_t)NATOM * NTC * 2);
  // per-block weight scratch (bf16, fragment-major packed)
  bf16* w_sigA  = (bf16*)carve(CC * 2);
  bf16* w_skipA = (bf16*)carve(CC * 2);
  bf16* w_q     = (bf16*)carve(CC * 2);
  bf16* w_k     = (bf16*)carve(CC * 2);
  bf16* w_v     = (bf16*)carve(CC * 2);
  bf16* w_g     = (bf16*)carve(CC * 2);
  bf16* w_o     = (bf16*)carve(CC * 2);
  bf16* w_s     = (bf16*)carve(CC * 2);
  bf16* w_sigT  = (bf16*)carve(CC * 2);
  bf16* w_skipT = (bf16*)carve(CC * 2);
  bf16* w_sT    = (bf16*)carve(CC * 2);
  bf16* w_1     = (bf16*)carve((size_t)CQ * NTC * 2);
  bf16* w_2     = (bf16*)carve((size_t)CQ * NTC * 2);
  bf16* w_oT    = (bf16*)carve((size_t)NTC * CQ * 2);

  auto cdiv = [](int a, int b) { return (a + b - 1) / b; };
  auto packb = [&](const float* src, bf16* dst, int K, int N) {
    pack_b_kernel<<<cdiv(K * N, 256), 256, 0, stream>>>(src, dst, K, N);
  };
  const int gemm_blocks_128 = ((NATOM / 16) * (CQ / 16)) / 8;   // 128
  const int gemm_blocks_256 = ((NATOM / 16) * (NTC / 16)) / 8;  // 256

  // ---- one-time prep ----
  hipMemcpyAsync(a_cur, ql, NC * 4, hipMemcpyDeviceToDevice, stream);
  ln_kernel<<<NATOM, CQ, 0, stream>>>(cl, lncl);
  cvt_bf16_kernel<<<cdiv((int)NC, 256), 256, 0, stream>>>(cl, cl_bf, (int)NC);
  zero_vt_pad_kernel<<<cdiv(CQ * 2 * VPAD, 256), 256, 0, stream>>>(vt_bf);

  for (int i = 0; i < NBLK; ++i) {
    // weight pack (f32 [K,N] -> bf16 fragment-major)
    packb(at_sig_w  + i * CC, w_sigA,  CQ, CQ);
    packb(at_skip_w + i * CC, w_skipA, CQ, CQ);
    packb(at_wq     + i * CC, w_q,     CQ, CQ);
    packb(at_wk     + i * CC, w_k,     CQ, CQ);
    packb(at_wv     + i * CC, w_v,     CQ, CQ);
    packb(at_wg     + i * CC, w_g,     CQ, CQ);
    packb(at_wo     + i * CC, w_o,     CQ, CQ);
    packb(at_ws_    + i * CC, w_s,     CQ, CQ);
    packb(tr_sig_w  + i * CC, w_sigT,  CQ, CQ);
    packb(tr_skip_w + i * CC, w_skipT, CQ, CQ);
    packb(tr_ws_    + i * CC, w_sT,    CQ, CQ);
    packb(tr_w1 + (size_t)i * CQ * NTC, w_1, CQ, NTC);
    packb(tr_w2 + (size_t)i * CQ * NTC, w_2, CQ, NTC);
    packb(tr_wo + (size_t)i * NTC * CQ, w_oT, NTC, CQ);

    // LN(a)
    ln_kernel<<<NATOM, CQ, 0, stream>>>(a_cur, lna);

    // ---- AttentionPairBias ----
    scale_bf16_kernel<<<cdiv((int)NC, 256), 256, 0, stream>>>(lncl, at_sln_g + i * CQ,
                                                              sn_bf, (int)NC);
    adaln_dual_kernel<<<gemm_blocks_128, 256, 0, stream>>>(sn_bf, w_sigA, w_skipA,
                                                           at_sig_b + i * CQ, lna, x_bf);
    qkvg_kernel<<<gemm_blocks_128, 256, 0, stream>>>(x_bf, w_q, at_bq + i * CQ,
                                                     w_k, w_v, w_g,
                                                     q_bf, k_bf, vt_bf, gbuf);
    attn_kernel<<<dim3(NATOM / NQW, NHEAD), 128, 0, stream>>>(
        q_bf, k_bf, vt_bf, gbuf, plm, amask,
        at_zln_g + i * CZP, at_zln_b + i * CZP, at_wz + i * CZP * NHEAD, og_bf);
    gemm_kernel<EPI_PLAIN><<<gemm_blocks_128, 256, 0, stream>>>(
        og_bf, w_o, nullptr, nullptr, nullptr, attnpre, NATOM, CQ, CQ);
    gemm_kernel<EPI_SIGMUL><<<gemm_blocks_128, 256, 0, stream>>>(
        cl_bf, w_s, at_bs + i * CQ, attnpre, nullptr, attnout, NATOM, CQ, CQ);

    // ---- ConditionedTransitionBlock ----
    scale_bf16_kernel<<<cdiv((int)NC, 256), 256, 0, stream>>>(lncl, tr_sln_g + i * CQ,
                                                              sn_bf, (int)NC);
    adaln_dual_kernel<<<gemm_blocks_128, 256, 0, stream>>>(sn_bf, w_sigT, w_skipT,
                                                           tr_sig_b + i * CQ, lna, xt_bf);
    swiglu_dual_kernel<<<gemm_blocks_256, 256, 0, stream>>>(xt_bf, w_1, w_2, hid_bf);
    gemm_kernel<EPI_PLAIN><<<gemm_blocks_128, 256, 0, stream>>>(
        hid_bf, w_oT, nullptr, nullptr, nullptr, tpre, NATOM, CQ, NTC);
    // a = sigmoid(cl@tr_ws + tr_bs) * tpre + attn_out   (last block -> d_out)
    float* dst = (i == NBLK - 1) ? out : a_cur;
    gemm_kernel<EPI_SIGMUL_ADD><<<gemm_blocks_128, 256, 0, stream>>>(
        cl_bf, w_sT, tr_bs + i * CQ, tpre, attnout, dst, NATOM, CQ, CQ);
  }
}
